// MultiHeadAttention_11828339933425
// MI455X (gfx1250) — matrix operand order repaired and checked by reading, not compiled, not run on hardware
//
#include <hip/hip_runtime.h>
#include <hip/hip_bf16.h>

// ---------------------------------------------------------------------------
// Multi-head attention for gfx1250 (MI455X), wave32, WMMA bf16 path.
//   B=2, S=2048, D=1024, H=16, d=64
// Flash-attention streaming softmax avoids the 537MB score matrix; all GEMMs
// go through v_wmma_f32_16x16x32_bf16 with f32 accumulation. K/V tiles are
// staged into LDS with GLOBAL_LOAD_ASYNC_TO_LDS (ASYNCcnt) when the builtins
// exist, with a plain vector-copy fallback otherwise.
// ---------------------------------------------------------------------------

typedef __bf16 bf16_t;
typedef __attribute__((ext_vector_type(16))) __bf16 v16bf;
typedef __attribute__((ext_vector_type(8)))  float  v8f;

static constexpr int Bn = 2;
static constexpr int Sn = 2048;
static constexpr int Dn = 1024;
static constexpr int Hn = 16;
static constexpr int dn = 64;   // head dim

#if __has_builtin(__builtin_amdgcn_global_load_async_to_lds_b128) && \
    __has_builtin(__builtin_amdgcn_s_wait_asynccnt)
#define USE_ASYNC_LDS 1
// builtin prototype (from compiler diagnostics): pointers to 16B int vectors,
// global side in the device (AS1) address space, LDS side in AS3.
typedef int v4i __attribute__((vector_size(16)));
#define ASYNC_GPTR(p) ((__attribute__((address_space(1))) v4i*)(p))
#define ASYNC_LPTR(p) ((__attribute__((address_space(3))) v4i*)(p))
#else
#define USE_ASYNC_LDS 0
#endif

__device__ inline v8f wmma_bf16(v16bf a, v16bf b, v8f c) {
    // (neg_a, A, neg_b, B, c_mod, C, reuse_a, reuse_b)
    return __builtin_amdgcn_wmma_f32_16x16x32_bf16(false, a, false, b,
                                                   (short)0, c, false, false);
}

// ---------------------------------------------------------------------------
// Kernel 0: fp32 -> bf16 conversion (for Wo)
// ---------------------------------------------------------------------------
__global__ void cvt_bf16_kernel(const float* __restrict__ src,
                                bf16_t* __restrict__ dst, int n) {
    int i = blockIdx.x * blockDim.x + threadIdx.x;
    if (i < n) dst[i] = (bf16_t)src[i];
}

// ---------------------------------------------------------------------------
// Kernel 1: fused QKV projection.
//   grid = (B*H, S/128), block = 256 (8 waves). Wave w -> 16 rows.
//   q = (x @ Wq^T + bq) * 1/sqrt(d)   (scale folded in here)
//   Q/K/V stored bf16, layout [B,H,S,d].
// ---------------------------------------------------------------------------
__global__ void qkv_kernel(const float* __restrict__ emb,
                           const float* __restrict__ Wq,
                           const float* __restrict__ Wk,
                           const float* __restrict__ Wv,
                           const float* __restrict__ bq,
                           const float* __restrict__ bk,
                           const float* __restrict__ bv,
                           bf16_t* __restrict__ Qw,
                           bf16_t* __restrict__ Kw,
                           bf16_t* __restrict__ Vw) {
    const int bh   = blockIdx.x;            // b*H + h
    const int b    = bh / Hn;
    const int h    = bh % Hn;
    const int wave = threadIdx.x >> 5;
    const int lane = threadIdx.x & 31;
    const int mrow = lane & 15;
    const int hi   = lane >> 4;
    const int r0   = blockIdx.y * 128 + wave * 16;   // first row of this wave

    // A-fragments: 16 rows of x_h, K-dim = head-dim (2 chunks of 32)
    v16bf xa[2];
#pragma unroll
    for (int kc = 0; kc < 2; ++kc) {
        const float* src = emb + ((size_t)b * Sn + r0 + mrow) * Dn + h * dn + kc * 32;
        const int kb = hi * 8;
#pragma unroll
        for (int e = 0; e < 8; ++e) {
            xa[kc][e]     = (bf16_t)src[kb + e];
            xa[kc][8 + e] = (bf16_t)src[kb + 16 + e];
        }
    }

    const float* Wmat[3] = { Wq + (size_t)h * dn * dn,
                             Wk + (size_t)h * dn * dn,
                             Wv + (size_t)h * dn * dn };
    const float* bias[3] = { bq + h * dn, bk + h * dn, bv + h * dn };
    bf16_t* outp[3] = { Qw + ((size_t)bh * Sn + r0) * dn,
                        Kw + ((size_t)bh * Sn + r0) * dn,
                        Vw + ((size_t)bh * Sn + r0) * dn };

#pragma unroll
    for (int p = 0; p < 3; ++p) {
        const float scl = (p == 0) ? 0.125f : 1.0f;   // 1/sqrt(64) on Q
#pragma unroll
        for (int nt = 0; nt < 4; ++nt) {
            const float bval = bias[p][nt * 16 + mrow];   // column = lane&15
            v8f acc;
#pragma unroll
            for (int r = 0; r < 8; ++r) acc[r] = bval;
#pragma unroll
            for (int kc = 0; kc < 2; ++kc) {
                // B[k,n] = W[n_global, k_global]  (y = x @ W^T)
                v16bf bw;
#pragma unroll
                for (int n = 0; n < 16; ++n)
                    bw[n] = (bf16_t)Wmat[p][(size_t)(nt * 16 + mrow) * dn + kc * 32 + hi * 8 + (n & 7) + (n >> 3) * 16];
                acc = wmma_bf16(xa[kc], bw, acc);
            }
#pragma unroll
            for (int r = 0; r < 8; ++r) {
                const int row = r + 8 * hi;
                outp[p][(size_t)row * dn + nt * 16 + mrow] = (bf16_t)(acc[r] * scl);
            }
        }
    }
}

// ---------------------------------------------------------------------------
// Kernel 2: flash attention per (b,h).
//   grid = (B*H, S/128), block = 256 (8 waves); wave owns 16 query rows.
//   K/V tiles of 64 keys staged in LDS via async-to-LDS loads; P restaged via
//   per-wave LDS to get A-fragment layout for P@V. Online softmax state per
//   accumulator row; next tile prefetched while current tile computes.
// ---------------------------------------------------------------------------
__global__ void attn_kernel(const bf16_t* __restrict__ Qw,
                            const bf16_t* __restrict__ Kw,
                            const bf16_t* __restrict__ Vw,
                            bf16_t* __restrict__ Ow) {
    __shared__ alignas(16) bf16_t Kt[64][72];
    __shared__ alignas(16) bf16_t Vt[64][72];
    __shared__ alignas(16) bf16_t Pt[8][16][72];

    const int bh   = blockIdx.x;
    const int b    = bh >> 4;
    const int h    = bh & 15;
    const int wave = threadIdx.x >> 5;
    const int lane = threadIdx.x & 31;
    const int mrow = lane & 15;
    const int hi   = lane >> 4;
    const size_t base = (size_t)bh * Sn * dn;
    const int r0   = blockIdx.y * 128 + wave * 16;

    // Q A-fragments (held in registers for the whole key loop)
    v16bf qa[2];
#pragma unroll
    for (int kc = 0; kc < 2; ++kc) {
        const bf16_t* src = Qw + base + (size_t)(r0 + mrow) * dn + kc * 32;
        const int kb = hi * 8;
#pragma unroll
        for (int e = 0; e < 8; ++e) {
            qa[kc][e]     = src[kb + e];
            qa[kc][8 + e] = src[kb + 16 + e];
        }
    }

    float mi[8], li[8];
    v8f oacc[4];
#pragma unroll
    for (int r = 0; r < 8; ++r) { mi[r] = -3.0e38f; li[r] = 0.0f; }
#pragma unroll
    for (int nt = 0; nt < 4; ++nt)
#pragma unroll
        for (int r = 0; r < 8; ++r) oacc[nt][r] = 0.0f;

    const int row = threadIdx.x >> 2;       // 0..63  cooperative-load row
    const int col = threadIdx.x & 3;        // 0..3   16-element (32B) chunk

    for (int kt = 0; kt < Sn / 64; ++kt) {
        __syncthreads();   // previous iteration's Vt/Pt reads complete

        // ---- stage K/V tile (64 keys x 64 dims, bf16): direct global->LDS
        {
            const bf16_t* gk = Kw + base + (size_t)(kt * 64 + row) * dn + col * 16;
            const bf16_t* gv = Vw + base + (size_t)(kt * 64 + row) * dn + col * 16;
#if USE_ASYNC_LDS
            __builtin_amdgcn_global_load_async_to_lds_b128(
                ASYNC_GPTR(gk),       ASYNC_LPTR(&Kt[row][col * 16]),     0, 0);
            __builtin_amdgcn_global_load_async_to_lds_b128(
                ASYNC_GPTR(gk + 8),   ASYNC_LPTR(&Kt[row][col * 16 + 8]), 0, 0);
            __builtin_amdgcn_global_load_async_to_lds_b128(
                ASYNC_GPTR(gv),       ASYNC_LPTR(&Vt[row][col * 16]),     0, 0);
            __builtin_amdgcn_global_load_async_to_lds_b128(
                ASYNC_GPTR(gv + 8),   ASYNC_LPTR(&Vt[row][col * 16 + 8]), 0, 0);
#else
            const uint4* gk4 = reinterpret_cast<const uint4*>(gk);
            const uint4* gv4 = reinterpret_cast<const uint4*>(gv);
            uint4* dk = reinterpret_cast<uint4*>(&Kt[row][col * 16]);
            uint4* dv = reinterpret_cast<uint4*>(&Vt[row][col * 16]);
            dk[0] = gk4[0]; dk[1] = gk4[1];
            dv[0] = gv4[0]; dv[1] = gv4[1];
#endif
            // prefetch next tile into cache while this tile is consumed
            if (kt + 1 < Sn / 64) {
                __builtin_prefetch(gk + dn * 64, 0, 3);
                __builtin_prefetch(gv + dn * 64, 0, 3);
            }
        }
#if USE_ASYNC_LDS
        __builtin_amdgcn_s_wait_asynccnt(0);
#endif
        __syncthreads();

        // ---- scores: S = Q @ K^T  (16 q-rows x 64 keys), 8 WMMAs
        v8f sacc[4];
#pragma unroll
        for (int nt = 0; nt < 4; ++nt)
#pragma unroll
            for (int r = 0; r < 8; ++r) sacc[nt][r] = 0.0f;
#pragma unroll
        for (int nt = 0; nt < 4; ++nt)
#pragma unroll
            for (int kc = 0; kc < 2; ++kc) {
                v16bf bw;   // B[e, t] = K[t, e]
#pragma unroll
                for (int n = 0; n < 16; ++n)
                    bw[n] = Kt[nt * 16 + mrow][kc * 32 + hi * 8 + (n & 7) + (n >> 3) * 16];
                sacc[nt] = wmma_bf16(qa[kc], bw, sacc[nt]);
            }

        // ---- online softmax (row reductions across 16-lane groups)
        float mnew[8], scalef[8];
#pragma unroll
        for (int r = 0; r < 8; ++r) {
            float mx = fmaxf(fmaxf(sacc[0][r], sacc[1][r]),
                             fmaxf(sacc[2][r], sacc[3][r]));
#pragma unroll
            for (int sh = 1; sh < 16; sh <<= 1)
                mx = fmaxf(mx, __shfl_xor(mx, sh, 32));
            mnew[r]   = fmaxf(mi[r], mx);
            scalef[r] = __expf(mi[r] - mnew[r]);
        }
#pragma unroll
        for (int nt = 0; nt < 4; ++nt)
#pragma unroll
            for (int r = 0; r < 8; ++r) {
                float p = __expf(sacc[nt][r] - mnew[r]);
                sacc[nt][r] = p;
                Pt[wave][r + 8 * hi][nt * 16 + mrow] = (bf16_t)p;
            }
#pragma unroll
        for (int r = 0; r < 8; ++r) {
            float rs = sacc[0][r] + sacc[1][r] + sacc[2][r] + sacc[3][r];
#pragma unroll
            for (int sh = 1; sh < 16; sh <<= 1)
                rs += __shfl_xor(rs, sh, 32);
            li[r] = li[r] * scalef[r] + rs;
            mi[r] = mnew[r];
        }
#pragma unroll
        for (int nt = 0; nt < 4; ++nt)
#pragma unroll
            for (int r = 0; r < 8; ++r) oacc[nt][r] *= scalef[r];

        __syncthreads();   // P visible in LDS

        // ---- O += P @ V  (16 x 64), 8 WMMAs
#pragma unroll
        for (int kc = 0; kc < 2; ++kc) {
            v16bf pa;
            const int kb = kc * 32 + hi * 8;
#pragma unroll
            for (int e = 0; e < 8; ++e) {
                pa[e]     = Pt[wave][mrow][kb + e];
                pa[8 + e] = Pt[wave][mrow][kb + 16 + e];
            }
#pragma unroll
            for (int nt = 0; nt < 4; ++nt) {
                v16bf bw;   // B[t, e] = V[t, e]
#pragma unroll
                for (int n = 0; n < 16; ++n)
                    bw[n] = Vt[kc * 32 + hi * 8 + (n & 7) + (n >> 3) * 16][nt * 16 + mrow];
                oacc[nt] = wmma_bf16(pa, bw, oacc[nt]);
            }
        }
    }

    // ---- normalize and store (layout [B, S, D], bf16 for final GEMM)
#pragma unroll
    for (int nt = 0; nt < 4; ++nt)
#pragma unroll
        for (int r = 0; r < 8; ++r) {
            const int srow = r0 + r + 8 * hi;
            Ow[((size_t)b * Sn + srow) * Dn + h * dn + nt * 16 + mrow] =
                (bf16_t)(oacc[nt][r] / li[r]);
        }
}

// ---------------------------------------------------------------------------
// Kernel 3: output projection  out = O @ Wo^T + bo  (fp32 result)
//   M = B*S = 4096, N = K = 1024
//   grid = (N/64, M/128), block = 256; wave -> 16 rows x 64 cols.
// ---------------------------------------------------------------------------
__global__ void oproj_kernel(const bf16_t* __restrict__ Ow,
                             const bf16_t* __restrict__ Wob,
                             const float* __restrict__ bo,
                             float* __restrict__ out) {
    const int wave = threadIdx.x >> 5;
    const int lane = threadIdx.x & 31;
    const int mrow = lane & 15;
    const int hi   = lane >> 4;
    const int m0   = blockIdx.y * 128 + wave * 16;
    const int n0   = blockIdx.x * 64;

    v8f acc[4];
#pragma unroll
    for (int nt = 0; nt < 4; ++nt) {
        const float bval = bo[n0 + nt * 16 + mrow];
#pragma unroll
        for (int r = 0; r < 8; ++r) acc[nt][r] = bval;
    }

    for (int k0 = 0; k0 < Dn; k0 += 32) {
        v16bf oa;
        const bf16_t* src = Ow + (size_t)(m0 + mrow) * Dn + k0;
        const int kb = hi * 8;
#pragma unroll
        for (int e = 0; e < 8; ++e) {
            oa[e]     = src[kb + e];
            oa[8 + e] = src[kb + 16 + e];
        }
#pragma unroll
        for (int nt = 0; nt < 4; ++nt) {
            v16bf bw;   // B[k,n] = Wo[n, k]
#pragma unroll
            for (int n = 0; n < 16; ++n)
                bw[n] = Wob[(size_t)(n0 + nt * 16 + mrow) * Dn + k0 + hi * 8 + (n & 7) + (n >> 3) * 16];
            acc[nt] = wmma_bf16(oa, bw, acc[nt]);
        }
    }

#pragma unroll
    for (int nt = 0; nt < 4; ++nt)
#pragma unroll
        for (int r = 0; r < 8; ++r)
            out[(size_t)(m0 + r + 8 * hi) * Dn + n0 + nt * 16 + mrow] = acc[nt][r];
}

// ---------------------------------------------------------------------------
// Host-side launcher
// ---------------------------------------------------------------------------
extern "C" void kernel_launch(void* const* d_in, const int* in_sizes, int n_in,
                              void* d_out, int out_size, void* d_ws, size_t ws_size,
                              hipStream_t stream) {
    const float* emb = (const float*)d_in[0];
    const float* Wq  = (const float*)d_in[1];
    const float* Wk  = (const float*)d_in[2];
    const float* Wv  = (const float*)d_in[3];
    const float* bq  = (const float*)d_in[4];
    const float* bk  = (const float*)d_in[5];
    const float* bv  = (const float*)d_in[6];
    const float* Wo  = (const float*)d_in[7];
    const float* bo  = (const float*)d_in[8];
    float* out = (float*)d_out;

    // workspace layout (bf16 buffers, all 32B aligned)
    const size_t nQKV = (size_t)Bn * Hn * Sn * dn;   // 4,194,304 elements
    const size_t nO   = (size_t)Bn * Sn * Dn;        // 4,194,304 elements
    bf16_t* Qw  = (bf16_t*)d_ws;
    bf16_t* Kw  = Qw + nQKV;
    bf16_t* Vw  = Kw + nQKV;
    bf16_t* Ow  = Vw + nQKV;
    bf16_t* Wob = Ow + nO;
    // total = (3*nQKV + nO + D*D) * 2 bytes ≈ 35.7 MB

    cvt_bf16_kernel<<<(Dn * Dn + 255) / 256, 256, 0, stream>>>(Wo, Wob, Dn * Dn);

    qkv_kernel<<<dim3(Bn * Hn, Sn / 128), 256, 0, stream>>>(
        emb, Wq, Wk, Wv, bq, bk, bv, Qw, Kw, Vw);

    attn_kernel<<<dim3(Bn * Hn, Sn / 128), 256, 0, stream>>>(Qw, Kw, Vw, Ow);

    oproj_kernel<<<dim3(Dn / 64, (Bn * Sn) / 128), 256, 0, stream>>>(
        Ow, Wob, bo, out);
}
